// ScaledDotProductAttention_3435973837236
// MI455X (gfx1250) — matrix operand order repaired and checked by reading, not compiled, not run on hardware
//
#include <hip/hip_runtime.h>
#include <math.h>

typedef __attribute__((ext_vector_type(16))) _Float16 v16h;
typedef __attribute__((ext_vector_type(8)))  _Float16 v8h;
typedef __attribute__((ext_vector_type(4)))  _Float16 v4h;
typedef __attribute__((ext_vector_type(8)))  float    v8f;
typedef int i32x4 __attribute__((vector_size(16)));

#define SEQ_L 1024
#define DIM   64
#define BQ    64
#define BK    64
#define KSTR  72          // padded LDS row stride (halves) = 144B, 16B aligned
#define NT    (SEQ_L / BK)
#define INV_T 0.125f      // 1/sqrt(64)

#if defined(__gfx1250__) && __has_builtin(__builtin_amdgcn_global_load_async_to_lds_b128)
#define HAS_ASYNC_LDS 1
#else
#define HAS_ASYNC_LDS 0
#endif

#if HAS_ASYNC_LDS
typedef __attribute__((address_space(1))) i32x4 gas_i32x4;
typedef __attribute__((address_space(3))) i32x4 las_i32x4;
#endif

static __device__ inline v16h cat16(v8h lo, v8h hi) {
    return __builtin_shufflevector(lo, hi, 0,1,2,3,4,5,6,7,8,9,10,11,12,13,14,15);
}

// 16-byte global -> LDS copy: async DMA path on gfx1250, register copy otherwise.
static __device__ inline void cp16(const _Float16* g, _Float16* l) {
#if HAS_ASYNC_LDS
    __builtin_amdgcn_global_load_async_to_lds_b128(
        (gas_i32x4*)g, (las_i32x4*)l, 0, 0);
#else
    *(v8h*)l = *(const v8h*)g;
#endif
}

template <int N>
static __device__ inline void wait_async_le() {
#if HAS_ASYNC_LDS
#if __has_builtin(__builtin_amdgcn_s_wait_asynccnt)
    __builtin_amdgcn_s_wait_asynccnt((short)N);
#else
    asm volatile("s_wait_asynccnt %0" :: "i"(N) : "memory");
#endif
#endif
}

// xor-shuffle within 16-lane half-groups (wave32): VALU permlane16 if available,
// else generic shuffle (which lowers to ds_bpermute).
template <int M, unsigned S0, unsigned S1>
static __device__ inline float xor16(float v) {
#if defined(__gfx1250__) && __has_builtin(__builtin_amdgcn_permlane16)
    unsigned u = __float_as_uint(v);
    return __uint_as_float(__builtin_amdgcn_permlane16(u, u, S0, S1, false, false));
#else
    return __shfl_xor(v, M, 32);
#endif
}
#define XOR16_1(x)  xor16<1, 0x67452301u, 0xEFCDAB89u>(x)
#define XOR16_2(x)  xor16<2, 0x54761032u, 0xDCFE98BAu>(x)
#define XOR16_4(x)  xor16<4, 0x32107654u, 0xBA98FEDCu>(x)
#define XOR16_8(x)  xor16<8, 0xFEDCBA98u, 0x76543210u>(x)

// ---------------------------------------------------------------------------
// Phase 1a: fp32 -> f16 elementwise (for Q and K)
// ---------------------------------------------------------------------------
__global__ __launch_bounds__(256)
void cvt_f16_kernel(const float* __restrict__ in, _Float16* __restrict__ out, int n4) {
    const int i = blockIdx.x * 256 + threadIdx.x;
    if (i < n4) {
        const float4 x = ((const float4*)in)[i];
        v4h h = { (_Float16)x.x, (_Float16)x.y, (_Float16)x.z, (_Float16)x.w };
        ((v4h*)out)[i] = h;
    }
}

// ---------------------------------------------------------------------------
// Phase 1b: V [B][L][D] fp32 -> Vt [B][D][L] f16 (64x64 LDS tile transpose)
// ---------------------------------------------------------------------------
__global__ __launch_bounds__(256)
void transpose_v_kernel(const float* __restrict__ v, _Float16* __restrict__ vt) {
    __shared__ _Float16 t[DIM][BK + 8];         // t[d][l]
    const int tid = threadIdx.x;
    const int b   = blockIdx.y;
    const int l0  = blockIdx.x * 64;
    const float* src = v + (size_t)b * SEQ_L * DIM + (size_t)l0 * DIM;
    const int c4 = tid & 15, r0 = tid >> 4;     // 16 float4 cols x 16 rows/pass
    #pragma unroll
    for (int p = 0; p < 4; ++p) {
        const int l = p * 16 + r0;
        const float4 x = ((const float4*)(src + (size_t)l * DIM))[c4];
        const int d = c4 * 4;
        t[d + 0][l] = (_Float16)x.x;
        t[d + 1][l] = (_Float16)x.y;
        t[d + 2][l] = (_Float16)x.z;
        t[d + 3][l] = (_Float16)x.w;
    }
    __syncthreads();
    _Float16* dst = vt + (size_t)b * DIM * SEQ_L + l0;
    const int c8 = tid & 7, rr = tid >> 3;      // 8 chunks x 32 rows/pass
    #pragma unroll
    for (int p = 0; p < 2; ++p) {
        const int d = p * 32 + rr;
        *(v8h*)(dst + (size_t)d * SEQ_L + c8 * 8) = *(const v8h*)(&t[d][c8 * 8]);
    }
}

// ---------------------------------------------------------------------------
// Phase 2: flash attention, f16 WMMA, double-buffered async LDS tiles
// ---------------------------------------------------------------------------
__global__ __launch_bounds__(128)
void fattn_wmma_f16_kernel(const _Float16* __restrict__ qh,
                           const _Float16* __restrict__ kh,
                           const _Float16* __restrict__ vth,
                           float* __restrict__ out)
{
    __shared__ _Float16 sK [2][BK * KSTR];      // K tile  [kv][d]
    __shared__ _Float16 sVt[2][DIM * KSTR];     // V tile  [d][kv]
    __shared__ _Float16 sP [4 * 16 * KSTR];     // per-wave P scratch

    const int tid  = threadIdx.x;
    const int lane = tid & 31;
    const int wave = tid >> 5;
    const int hg   = lane >> 4;
    const int ln   = lane & 15;

    const int qt = blockIdx.x;
    const int b  = blockIdx.y;
    const size_t boff = (size_t)b * SEQ_L * DIM;
    const _Float16* qb = qh  + boff + (size_t)qt * BQ * DIM;
    const _Float16* kb = kh  + boff;
    const _Float16* vb = vth + boff;            // [d][L]

    // Q A-fragments straight from global (f16 rows are contiguous 128B)
    v16h aq[2];
    {
        const _Float16* qrow = qb + (size_t)(wave * 16 + ln) * DIM;
        #pragma unroll
        for (int ks = 0; ks < 2; ++ks)
            aq[ks] = cat16(*(const v8h*)(qrow + ks * 32 + hg * 8),
                           *(const v8h*)(qrow + ks * 32 + hg * 8 + 16));
    }

    v8f  o_acc[4];
    float m_r[8], l_r[8];
    #pragma unroll
    for (int t = 0; t < 4; ++t) o_acc[t] = (v8f){0,0,0,0,0,0,0,0};
    #pragma unroll
    for (int g = 0; g < 8; ++g) { m_r[g] = -INFINITY; l_r[g] = 0.0f; }

    // 512 16B-chunk tasks per matrix per tile; 4 per thread per matrix.
    auto issue_tile = [&](int buf, int t) {
        #pragma unroll
        for (int p = 0; p < 4; ++p) {
            const int task = p * 128 + tid;     // 0..511
            const int r = task >> 3, c = task & 7;
            cp16(kb + (size_t)t * BK * DIM + (size_t)task * 8,
                 &sK[buf][r * KSTR + c * 8]);
            cp16(vb + (size_t)r * SEQ_L + t * BK + c * 8,
                 &sVt[buf][r * KSTR + c * 8]);
        }
    };

    issue_tile(0, 0);

    for (int kt = 0; kt < NT; ++kt) {
        const int buf = kt & 1;
        if (kt + 1 < NT) { issue_tile(buf ^ 1, kt + 1); wait_async_le<8>(); }
        else             { wait_async_le<0>(); }
        __syncthreads();

        // ---- S = (Q K^T) * inv_temp ----
        v8f s[4];
        #pragma unroll
        for (int t = 0; t < 4; ++t) {
            v8f c = (v8f){0,0,0,0,0,0,0,0};
            #pragma unroll
            for (int ks = 0; ks < 2; ++ks) {
                const _Float16* src = &sK[buf][(t * 16 + ln) * KSTR + ks * 32 + hg * 8];
                v16h bk = cat16(*(const v8h*)(src), *(const v8h*)(src + 16));
                c = __builtin_amdgcn_wmma_f32_16x16x32_f16(
                        false, aq[ks], false, bk, (short)0, c, false, false);
            }
            s[t] = c * INV_T;
        }

        // ---- online softmax (row stats within 16-lane half-group) ----
        float alpha[8];
        #pragma unroll
        for (int g = 0; g < 8; ++g) {
            float rm = fmaxf(fmaxf(s[0][g], s[1][g]), fmaxf(s[2][g], s[3][g]));
            rm = fmaxf(rm, XOR16_1(rm));
            rm = fmaxf(rm, XOR16_2(rm));
            rm = fmaxf(rm, XOR16_4(rm));
            rm = fmaxf(rm, XOR16_8(rm));
            const float mn = fmaxf(m_r[g], rm);
            const float a  = __expf(m_r[g] - mn);
            float rs = 0.0f;
            #pragma unroll
            for (int t = 0; t < 4; ++t) {
                const float p = __expf(s[t][g] - mn);
                s[t][g] = p;
                rs += p;
            }
            rs += XOR16_1(rs);
            rs += XOR16_2(rs);
            rs += XOR16_4(rs);
            rs += XOR16_8(rs);
            l_r[g] = l_r[g] * a + rs;
            m_r[g] = mn;
            alpha[g] = a;
        }
        #pragma unroll
        for (int t = 0; t < 4; ++t)
            #pragma unroll
            for (int g = 0; g < 8; ++g)
                o_acc[t][g] *= alpha[g];

        // ---- rotate P: C layout -> per-wave LDS -> A layout ----
        _Float16* pw = &sP[wave * 16 * KSTR];
        #pragma unroll
        for (int t = 0; t < 4; ++t)
            #pragma unroll
            for (int g = 0; g < 8; ++g)
                pw[(g + 8 * hg) * KSTR + t * 16 + ln] = (_Float16)s[t][g];

        v16h ap[2];
        #pragma unroll
        for (int ks = 0; ks < 2; ++ks) {
            const _Float16* src = &pw[ln * KSTR + ks * 32 + hg * 8];
            ap[ks] = cat16(*(const v8h*)(src), *(const v8h*)(src + 16));
        }

        // ---- O += P @ V ----
        #pragma unroll
        for (int t = 0; t < 4; ++t) {
            #pragma unroll
            for (int ks = 0; ks < 2; ++ks) {
                const _Float16* src = &sVt[buf][(t * 16 + ln) * KSTR + ks * 32 + hg * 8];
                v16h bv = cat16(*(const v8h*)(src), *(const v8h*)(src + 16));
                o_acc[t] = __builtin_amdgcn_wmma_f32_16x16x32_f16(
                               false, ap[ks], false, bv, (short)0, o_acc[t], false, false);
            }
        }
        __syncthreads();
    }

    // ---- normalize and store (C layout: M = g + 8*hg, col = t*16 + ln) ----
    float* og = out + boff + (size_t)qt * BQ * DIM;
    #pragma unroll
    for (int t = 0; t < 4; ++t) {
        #pragma unroll
        for (int g = 0; g < 8; ++g) {
            const int M = wave * 16 + hg * 8 + g;
            og[(size_t)M * DIM + t * 16 + ln] = o_acc[t][g] / l_r[g];
        }
    }
}

// ---------------------------------------------------------------------------
// Fallback: single-pass fp32-input kernel (used only if workspace too small)
// ---------------------------------------------------------------------------
__global__ __launch_bounds__(128)
void fattn_wmma_kernel(const float* __restrict__ q, const float* __restrict__ k,
                       const float* __restrict__ v, float* __restrict__ out)
{
    __shared__ _Float16 sQ [BQ * KSTR];
    __shared__ _Float16 sK [BK * KSTR];
    __shared__ _Float16 sVt[DIM * KSTR];
    __shared__ _Float16 sP [4 * 16 * KSTR];

    const int tid  = threadIdx.x;
    const int lane = tid & 31;
    const int wave = tid >> 5;
    const int hg   = lane >> 4;
    const int ln   = lane & 15;

    const int qt = blockIdx.x;
    const int b  = blockIdx.y;
    const size_t base = (size_t)b * SEQ_L * DIM;
    const float* qg = q + base + (size_t)qt * BQ * DIM;

    {
        const int c4 = tid & 15, r0 = tid >> 4;
        #pragma unroll
        for (int p = 0; p < 8; ++p) {
            const int r = p * 8 + r0;
            const float4 x = ((const float4*)(qg + (size_t)r * DIM))[c4];
            _Float16* dst = &sQ[r * KSTR + c4 * 4];
            dst[0] = (_Float16)x.x; dst[1] = (_Float16)x.y;
            dst[2] = (_Float16)x.z; dst[3] = (_Float16)x.w;
        }
    }
    __syncthreads();

    v16h aq[2];
    {
        const int row = wave * 16 + ln;
        #pragma unroll
        for (int ks = 0; ks < 2; ++ks) {
            const _Float16* src = &sQ[row * KSTR + ks * 32 + hg * 8];
            aq[ks] = cat16(*(const v8h*)(src), *(const v8h*)(src + 16));
        }
    }

    v8f  o_acc[4];
    float m_r[8], l_r[8];
    #pragma unroll
    for (int t = 0; t < 4; ++t) o_acc[t] = (v8f){0,0,0,0,0,0,0,0};
    #pragma unroll
    for (int g = 0; g < 8; ++g) { m_r[g] = -INFINITY; l_r[g] = 0.0f; }

    const float* kg = k + base;
    const float* vg = v + base;

    for (int kt = 0; kt < NT; ++kt) {
        __syncthreads();
        {
            const int c4 = tid & 15, r0 = tid >> 4;
            const float* kgt = kg + (size_t)kt * BK * DIM;
            const float* vgt = vg + (size_t)kt * BK * DIM;
            #pragma unroll
            for (int p = 0; p < 8; ++p) {
                const int r = p * 8 + r0;
                const float4 x = ((const float4*)(kgt + (size_t)r * DIM))[c4];
                _Float16* dst = &sK[r * KSTR + c4 * 4];
                dst[0] = (_Float16)x.x; dst[1] = (_Float16)x.y;
                dst[2] = (_Float16)x.z; dst[3] = (_Float16)x.w;
                const float4 y = ((const float4*)(vgt + (size_t)r * DIM))[c4];
                const int c = c4 * 4;
                sVt[(c + 0) * KSTR + r] = (_Float16)y.x;
                sVt[(c + 1) * KSTR + r] = (_Float16)y.y;
                sVt[(c + 2) * KSTR + r] = (_Float16)y.z;
                sVt[(c + 3) * KSTR + r] = (_Float16)y.w;
            }
        }
        __syncthreads();

        v8f s[4];
        #pragma unroll
        for (int t = 0; t < 4; ++t) {
            v8f c = (v8f){0,0,0,0,0,0,0,0};
            #pragma unroll
            for (int ks = 0; ks < 2; ++ks) {
                const _Float16* src = &sK[(t * 16 + ln) * KSTR + ks * 32 + hg * 8];
                v16h bk = cat16(*(const v8h*)(src), *(const v8h*)(src + 16));
                c = __builtin_amdgcn_wmma_f32_16x16x32_f16(
                        false, aq[ks], false, bk, (short)0, c, false, false);
            }
            s[t] = c * INV_T;
        }

        float alpha[8];
        #pragma unroll
        for (int g = 0; g < 8; ++g) {
            float rm = fmaxf(fmaxf(s[0][g], s[1][g]), fmaxf(s[2][g], s[3][g]));
            rm = fmaxf(rm, XOR16_1(rm));
            rm = fmaxf(rm, XOR16_2(rm));
            rm = fmaxf(rm, XOR16_4(rm));
            rm = fmaxf(rm, XOR16_8(rm));
            const float mn = fmaxf(m_r[g], rm);
            const float a  = __expf(m_r[g] - mn);
            float rs = 0.0f;
            #pragma unroll
            for (int t = 0; t < 4; ++t) {
                const float p = __expf(s[t][g] - mn);
                s[t][g] = p;
                rs += p;
            }
            rs += XOR16_1(rs);
            rs += XOR16_2(rs);
            rs += XOR16_4(rs);
            rs += XOR16_8(rs);
            l_r[g] = l_r[g] * a + rs;
            m_r[g] = mn;
            alpha[g] = a;
        }
        #pragma unroll
        for (int t = 0; t < 4; ++t)
            #pragma unroll
            for (int g = 0; g < 8; ++g)
                o_acc[t][g] *= alpha[g];

        _Float16* pw = &sP[wave * 16 * KSTR];
        #pragma unroll
        for (int t = 0; t < 4; ++t)
            #pragma unroll
            for (int g = 0; g < 8; ++g)
                pw[(g + 8 * hg) * KSTR + t * 16 + ln] = (_Float16)s[t][g];

        v16h ap[2];
        #pragma unroll
        for (int ks = 0; ks < 2; ++ks) {
            const _Float16* src = &pw[ln * KSTR + ks * 32 + hg * 8];
            ap[ks] = cat16(*(const v8h*)(src), *(const v8h*)(src + 16));
        }

        #pragma unroll
        for (int t = 0; t < 4; ++t) {
            #pragma unroll
            for (int ks = 0; ks < 2; ++ks) {
                const _Float16* src = &sVt[(t * 16 + ln) * KSTR + ks * 32 + hg * 8];
                v16h bv = cat16(*(const v8h*)(src), *(const v8h*)(src + 16));
                o_acc[t] = __builtin_amdgcn_wmma_f32_16x16x32_f16(
                               false, ap[ks], false, bv, (short)0, o_acc[t], false, false);
            }
        }
    }

    float* og = out + base + (size_t)qt * BQ * DIM;
    #pragma unroll
    for (int t = 0; t < 4; ++t) {
        #pragma unroll
        for (int g = 0; g < 8; ++g) {
            const int M = wave * 16 + hg * 8 + g;
            og[(size_t)M * DIM + t * 16 + ln] = o_acc[t][g] / l_r[g];
        }
    }
}

extern "C" void kernel_launch(void* const* d_in, const int* in_sizes, int n_in,
                              void* d_out, int out_size, void* d_ws, size_t ws_size,
                              hipStream_t stream) {
    const float* q = (const float*)d_in[0];
    const float* k = (const float*)d_in[1];
    const float* v = (const float*)d_in[2];
    float* out = (float*)d_out;
    const int B = in_sizes[0] / (SEQ_L * DIM);
    const size_t elems = (size_t)B * SEQ_L * DIM;
    const size_t need  = 3 * elems * 2;          // q,k,vt in f16

    if (d_ws != nullptr && ws_size >= need) {
        _Float16* qh = (_Float16*)d_ws;
        _Float16* kh = qh + elems;
        _Float16* vt = kh + elems;
        const int n4 = (int)(elems / 4);
        hipLaunchKernelGGL(cvt_f16_kernel, dim3((n4 + 255) / 256), dim3(256), 0, stream, q, qh, n4);
        hipLaunchKernelGGL(cvt_f16_kernel, dim3((n4 + 255) / 256), dim3(256), 0, stream, k, kh, n4);
        hipLaunchKernelGGL(transpose_v_kernel, dim3(SEQ_L / 64, B), dim3(256), 0, stream, v, vt);
        hipLaunchKernelGGL(fattn_wmma_f16_kernel, dim3(SEQ_L / BQ, B), dim3(128), 0, stream,
                           qh, kh, vt, out);
    } else {
        hipLaunchKernelGGL(fattn_wmma_kernel, dim3(SEQ_L / BQ, B), dim3(128), 0, stream,
                           q, k, v, out);
    }
}
